// ScaledDotProductAttention_4672924418742
// MI455X (gfx1250) — matrix operand order repaired and checked by reading, not compiled, not run on hardware
//
#include <hip/hip_runtime.h>
#include <hip/hip_bf16.h>

#define HEADS   8
#define DM      512
#define DK      64
#define SEQ     2048   // NQ == NK
#define BATCH   4

typedef __attribute__((ext_vector_type(16))) _Float16 v16h;
typedef __attribute__((ext_vector_type(8)))  _Float16 v8h;
typedef __attribute__((ext_vector_type(8)))  float    v8f;

union V16H { v16h v; v8h h[2]; _Float16 e[16]; };

// ---------------------------------------------------------------- WMMA helpers

__device__ __forceinline__ v8f wmma_f16(v16h a, v16h b, v8f c) {
    // D = A(16x32 f16) * B(32x16 f16) + C(16x16 f32)
    return __builtin_amdgcn_wmma_f32_16x16x32_f16(false, a, false, b,
                                                  (short)0, c, false, false);
}

// B-fragment (32x16, f16): lane L holds column n=L%16, elements e -> K = (L/16)*16 + e.
// Caller passes pointer already offset to (col, kbase + hi*16); 16 contiguous halves.
__device__ __forceinline__ v16h ldb16(const _Float16* p) {
    V16H u;
    u.h[0] = *(const v8h*)(p);
    u.h[1] = *(const v8h*)(p + 8);
    return u.v;
}

// A-fragment (16x32, f16): lane holds row M=L%16.
// lanes 0-15: e0..15 -> K 0..15
// lanes16-31: e0..15 -> K 16..31
__device__ __forceinline__ v16h lda16(const _Float16* row, int kk, int hi) {
    V16H u;
    u.h[0] = *(const v8h*)(row + kk + hi * 16);
    u.h[1] = *(const v8h*)(row + kk + 8 + hi * 16);
    return u.v;
}

// Same A-fragment but source is f32 (on-the-fly convert; data lives in L2).
__device__ __forceinline__ v16h lda32(const float* row, int kk, int hi) {
    V16H u;
    const float* c0 = row + kk + hi * 16;
    const float* c1 = row + kk + 8 + hi * 16;
#pragma unroll
    for (int i = 0; i < 8; ++i) {
        u.e[i]     = (_Float16)c0[i];
        u.e[8 + i] = (_Float16)c1[i];
    }
    return u.v;
}

// ---------------------------------------------------------------- cvt+transpose
// WT[n*K + k] = (f16) W[k*N + n]   (512x512, trivial traffic)
__global__ void cvt_transpose_f16(const float* __restrict__ W,
                                  _Float16* __restrict__ WT, int K, int N) {
    int idx = blockIdx.x * blockDim.x + threadIdx.x;
    if (idx >= K * N) return;
    int k = idx / N, n = idx % N;
    WT[(size_t)n * K + k] = (_Float16)W[(size_t)k * N + n];
}

// ---------------------------------------------------------------- GEMM (wave = 16x16 tile)
// MODE 0: A=f32 (M x K), out f16 -> (b,h,tok,d)      [Q,K projections]
// MODE 1: A=f32 (M x K), out f16 -> (b,h,d,tok)      [V projection, pre-transposed]
// MODE 2: A=f16 (M x K), out f32 -> (M x N) + bias   [output projection]
template <int MODE>
__global__ __launch_bounds__(256) void gemm_wmma(const void* __restrict__ Aptr,
                                                 const _Float16* __restrict__ BT,
                                                 const float* __restrict__ bias,
                                                 void* __restrict__ Out,
                                                 int M, int N, int K) {
    const int wave = threadIdx.x >> 5;
    const int lane = threadIdx.x & 31;
    const int n    = lane & 15;
    const int hi   = lane >> 4;

    const int tiles_n = N >> 4;
    const int tile    = blockIdx.x * 8 + wave;
    const int mt      = tile / tiles_n;
    const int nt      = tile % tiles_n;

    const _Float16* brow = BT + (size_t)(nt * 16 + n) * K;

    const float*    arow32 = (const float*)Aptr    + (size_t)(mt * 16 + n) * K;
    const _Float16* arow16 = (const _Float16*)Aptr + (size_t)(mt * 16 + n) * K;

    v8f acc = (v8f)0.0f;
#pragma unroll 4
    for (int kk = 0; kk < K; kk += 32) {
        v16h a;
        if constexpr (MODE == 2) a = lda16(arow16, kk, hi);
        else                     a = lda32(arow32, kk, hi);
        v16h b = ldb16(brow + kk + hi * 16);
        acc = wmma_f16(a, b, acc);
    }

    const int   nglob = nt * 16 + n;
    const float bb    = bias[nglob];
#pragma unroll
    for (int r = 0; r < 8; ++r) {
        const int   m   = mt * 16 + r + 8 * hi;
        const float val = acc[r] + bb;
        if constexpr (MODE == 2) {
            ((float*)Out)[(size_t)m * N + nglob] = val;
        } else {
            const int tok = m & (SEQ - 1);
            const int bi  = m >> 11;            // m / SEQ
            const int h   = nglob >> 6;         // / 64
            const int d   = nglob & 63;
            if constexpr (MODE == 0)
                ((_Float16*)Out)[(((size_t)bi * HEADS + h) * SEQ + tok) * DK + d] = (_Float16)val;
            else
                ((_Float16*)Out)[(((size_t)bi * HEADS + h) * DK + d) * SEQ + tok] = (_Float16)val;
        }
    }
}

// ---------------------------------------------------------------- fused flash attention
// wave = 16 query rows, streams 32 keys/iter; online softmax; never materializes att.
#define PSTRIDE 40   // LDS row stride in halves (80B: 16B aligned for b128 loads)

__global__ __launch_bounds__(256) void attn_fused(const _Float16* __restrict__ Qp,
                                                  const _Float16* __restrict__ Kp,
                                                  const _Float16* __restrict__ VpT,
                                                  const float* __restrict__ AW,
                                                  const unsigned char* __restrict__ Mask,
                                                  _Float16* __restrict__ OutH) {
    __shared__ __align__(16) _Float16 lds[8 * 16 * PSTRIDE];

    const int wave = threadIdx.x >> 5;
    const int lane = threadIdx.x & 31;
    const int n    = lane & 15;
    const int hi   = lane >> 4;

    const int wg = blockIdx.x * 8 + wave;      // 4096 total
    const int bh = wg >> 7;                    // / (SEQ/16)
    const int qt = wg & 127;
    const int b  = bh / HEADS;
    const int h  = bh % HEADS;

    const _Float16* Qb = Qp  + (size_t)bh * SEQ * DK;
    const _Float16* Kb = Kp  + (size_t)bh * SEQ * DK;
    const _Float16* Vb = VpT + (size_t)bh * DK * SEQ;
    const float*         AWb = AW   + (size_t)bh * SEQ * SEQ;
    const unsigned char* Mb  = Mask + (size_t)bh * SEQ * SEQ;
    _Float16* pl = &lds[wave * 16 * PSTRIDE];

    // Q A-fragments held in registers for the whole loop (16x64 -> two 16x32 frags)
    const _Float16* qrow = Qb + (size_t)(qt * 16 + n) * DK;
    const v16h qa0 = lda16(qrow, 0, hi);
    const v16h qa1 = lda16(qrow, 32, hi);

    float m_[8], l_[8];
    v8f acc[4];
#pragma unroll
    for (int r = 0; r < 8; ++r) { m_[r] = -1e30f; l_[r] = 0.0f; }
#pragma unroll
    for (int g = 0; g < 4; ++g) acc[g] = (v8f)0.0f;

    const float scale = 0.125f;   // 1/sqrt(64)

    for (int kt = 0; kt < SEQ / 32; ++kt) {
        const int key0 = kt * 32;

        // ================= issue ALL independent global loads up front =======
        // K B-fragments (4 x 2 x b128)
        const _Float16* kc0 = Kb + (size_t)(key0 + n) * DK;
        const _Float16* kc1 = Kb + (size_t)(key0 + 16 + n) * DK;
        const v16h kb00 = ldb16(kc0 + hi * 16);
        const v16h kb01 = ldb16(kc0 + 32 + hi * 16);
        const v16h kb10 = ldb16(kc1 + hi * 16);
        const v16h kb11 = ldb16(kc1 + 32 + hi * 16);

        // V B-fragments (4 x 2 x b128) — latency hidden behind softmax VALU
        v16h vb[4];
#pragma unroll
        for (int g = 0; g < 4; ++g)
            vb[g] = ldb16(Vb + (size_t)(g * 16 + n) * SEQ + key0 + hi * 16);

        // attention_weights + mask (the streaming 537MB/134MB tensors)
        float aw0[8], aw1[8];
        unsigned char mk0[8], mk1[8];
#pragma unroll
        for (int r = 0; r < 8; ++r) {
            const size_t off = (size_t)(qt * 16 + r + 8 * hi) * SEQ + key0 + n;
            aw0[r] = AWb[off];
            aw1[r] = AWb[off + 16];
            mk0[r] = Mb[off];
            mk1[r] = Mb[off + 16];
        }

        // prefetch next iteration's aw/mask cachelines (-> global_prefetch_b8)
        if (kt + 1 < SEQ / 32) {
            const size_t pf = (size_t)(qt * 16 + (lane & 7) + 8 * hi) * SEQ + key0 + 32;
            __builtin_prefetch(AWb + pf, 0, 1);
            __builtin_prefetch(Mb + pf, 0, 1);
        }

        // ================= S = Q K^T (two 16x16 column tiles) ================
        v8f s0 = (v8f)0.0f, s1 = (v8f)0.0f;
        s0 = wmma_f16(qa0, kb00, s0);
        s0 = wmma_f16(qa1, kb01, s0);
        s1 = wmma_f16(qa0, kb10, s1);
        s1 = wmma_f16(qa1, kb11, s1);

        // ---- scale * aw, mask -> -1e30 (branchless: cndmask + min, aw always used)
        float p0[8], p1[8];
#pragma unroll
        for (int r = 0; r < 8; ++r) {
            const float cap0 = mk0[r] ? -1e30f : 3.0e38f;
            const float cap1 = mk1[r] ? -1e30f : 3.0e38f;
            p0[r] = fminf(s0[r] * scale * aw0[r], cap0);
            p1[r] = fminf(s1[r] * scale * aw1[r], cap1);
        }

        // ---- online softmax (row stats replicated across each 16-lane half)
        float fac[8];
#pragma unroll
        for (int r = 0; r < 8; ++r) {
            float tm = fmaxf(p0[r], p1[r]);
#pragma unroll
            for (int off = 1; off < 16; off <<= 1)
                tm = fmaxf(tm, __shfl_xor(tm, off, 32));
            const float mnew = fmaxf(m_[r], tm);
            const float e0 = __expf(p0[r] - mnew);
            const float e1 = __expf(p1[r] - mnew);
            p0[r] = e0; p1[r] = e1;
            float ps = e0 + e1;
#pragma unroll
            for (int off = 1; off < 16; off <<= 1)
                ps += __shfl_xor(ps, off, 32);
            fac[r] = __expf(m_[r] - mnew);
            l_[r]  = l_[r] * fac[r] + ps;
            m_[r]  = mnew;
        }
#pragma unroll
        for (int g = 0; g < 4; ++g)
#pragma unroll
            for (int r = 0; r < 8; ++r) acc[g][r] *= fac[r];

        // ---- P (C-layout) -> LDS f16 row-major, reload as A-fragment
#pragma unroll
        for (int r = 0; r < 8; ++r) {
            const int row = r + 8 * hi;
            pl[row * PSTRIDE + n]      = (_Float16)p0[r];
            pl[row * PSTRIDE + 16 + n] = (_Float16)p1[r];
        }
        asm volatile("s_wait_dscnt 0x0" ::: "memory");   // wave-private LDS: counter wait suffices
        const v16h pa = lda16(pl + n * PSTRIDE, 0, hi);

        // ---- acc += P * V   (V fragments already resident)
#pragma unroll
        for (int g = 0; g < 4; ++g)
            acc[g] = wmma_f16(pa, vb[g], acc[g]);
    }

    // ---- normalize and store f16 attn-out (token-major for the final GEMM)
#pragma unroll
    for (int g = 0; g < 4; ++g)
#pragma unroll
        for (int r = 0; r < 8; ++r) {
            const int row = qt * 16 + r + 8 * hi;
            const float o = acc[g][r] / l_[r];
            OutH[((size_t)b * SEQ + row) * DM + h * DK + g * 16 + n] = (_Float16)o;
        }
}

// ---------------------------------------------------------------- launch

extern "C" void kernel_launch(void* const* d_in, const int* in_sizes, int n_in,
                              void* d_out, int out_size, void* d_ws, size_t ws_size,
                              hipStream_t stream) {
    const float* queries = (const float*)d_in[0];
    const float* keys    = (const float*)d_in[1];
    const float* values  = (const float*)d_in[2];
    const float* aw      = (const float*)d_in[3];
    const unsigned char* mask = (const unsigned char*)d_in[4];
    const float* Wq = (const float*)d_in[5];
    const float* bq = (const float*)d_in[6];
    const float* Wk = (const float*)d_in[7];
    const float* bk = (const float*)d_in[8];
    const float* Wv = (const float*)d_in[9];
    const float* bv = (const float*)d_in[10];
    const float* Wo = (const float*)d_in[11];
    const float* bo = (const float*)d_in[12];

    const size_t WSZ  = (size_t)DM * DM;            // 262144
    const size_t PSZ  = (size_t)BATCH * SEQ * DM;   // 4194304 (== B*H*SEQ*DK)

    _Float16* p = (_Float16*)d_ws;
    _Float16* WqT = p; p += WSZ;
    _Float16* WkT = p; p += WSZ;
    _Float16* WvT = p; p += WSZ;
    _Float16* WoT = p; p += WSZ;
    _Float16* Qp  = p; p += PSZ;
    _Float16* Kp  = p; p += PSZ;
    _Float16* VpT = p; p += PSZ;
    _Float16* AttO = p; p += PSZ;

    const int M = BATCH * SEQ;                      // 8192

    // weights -> f16, transposed (B-fragment wants contiguous K per column)
    const int tthreads = 256, tblocks = (int)((WSZ + 255) / 256);
    cvt_transpose_f16<<<tblocks, tthreads, 0, stream>>>(Wq, WqT, DM, DM);
    cvt_transpose_f16<<<tblocks, tthreads, 0, stream>>>(Wk, WkT, DM, DM);
    cvt_transpose_f16<<<tblocks, tthreads, 0, stream>>>(Wv, WvT, DM, DM);
    cvt_transpose_f16<<<tblocks, tthreads, 0, stream>>>(Wo, WoT, DM, DM);

    // projections: (8192x512) @ (512x512); 16384 tiles / 8 waves = 2048 blocks
    const int gblocks = (M / 16) * (DM / 16) / 8;
    gemm_wmma<0><<<gblocks, 256, 0, stream>>>(queries, WqT, bq, Qp,  M, DM, DM);
    gemm_wmma<0><<<gblocks, 256, 0, stream>>>(keys,    WkT, bk, Kp,  M, DM, DM);
    gemm_wmma<1><<<gblocks, 256, 0, stream>>>(values,  WvT, bv, VpT, M, DM, DM);

    // fused attention: 4096 q-tiles / 8 waves = 512 blocks
    attn_fused<<<512, 256, 0, stream>>>(Qp, Kp, VpT, aw, mask, AttO);

    // output projection -> f32 d_out
    gemm_wmma<2><<<gblocks, 256, 0, stream>>>(AttO, WoT, bo, d_out, M, DM, DM);
}
